// LegalMultiHeadAttention_53455162966324
// MI455X (gfx1250) — matrix operand order repaired and checked by reading, not compiled, not run on hardware
//
#include <hip/hip_runtime.h>

// ---- problem constants ----
#define H_     4
#define B_     2
#define S_     4096
#define DM_    1024
#define DK_    256
#define ROWS_  (B_ * S_)          // 8192

// ---- WMMA types ----
typedef __attribute__((ext_vector_type(16))) _Float16 v16h;
typedef __attribute__((ext_vector_type(8)))  _Float16 v8h;
typedef __attribute__((ext_vector_type(8)))  float    v8f;

// LDS tile row stride in halves: 32 (BK) + 8 pad -> 80 bytes, 16B aligned
#define TSTR 40

// ------------------------------------------------------------------
// Fragment loaders (layouts per CDNA5 ISA 7.12.2, wave32)
// A: 16x32 f16. lanes 0-15: m=lane, k in {0..7}U{16..23};
//               lanes 16-31: m=lane-16, k in {8..15}U{24..31}.
// ------------------------------------------------------------------
__device__ __forceinline__ v16h load_frag_a(const _Float16* tile, int mbase, int lane) {
    const int m  = mbase + (lane & 15);
    const int kb = (lane & 16) ? 8 : 0;
    const _Float16* p = tile + m * TSTR + kb;
    v8h lo = *(const v8h*)(p);        // k = kb .. kb+7
    v8h hi = *(const v8h*)(p + 16);   // k = kb+16 .. kb+23
    v16h f;
#pragma unroll
    for (int i = 0; i < 8; ++i) { f[i] = lo[i]; f[i + 8] = hi[i]; }
    return f;
}

// B stored transposed in LDS: Bt[n][k]. lanes 0-15: n=lane, k in {0..7}U{16..23};
// lanes 16-31: n=lane-16, k in {8..15}U{24..31} (same k order as A).
__device__ __forceinline__ v16h load_frag_b(const _Float16* tile, int nbase, int lane) {
    const int n  = nbase + (lane & 15);
    const int kb = (lane & 16) ? 8 : 0;
    const _Float16* p = tile + n * TSTR + kb;
    v8h lo = *(const v8h*)(p);
    v8h hi = *(const v8h*)(p + 16);
    v16h f;
#pragma unroll
    for (int i = 0; i < 8; ++i) { f[i] = lo[i]; f[i + 8] = hi[i]; }
    return f;
}

#define WMMA_F16(a, b, c) \
    __builtin_amdgcn_wmma_f32_16x16x32_f16(false, (a), false, (b), (short)0, (c), false, false)

__device__ __forceinline__ v8h cvt8(const float4 a, const float4 b) {
    v8h h;
    h[0] = (_Float16)a.x; h[1] = (_Float16)a.y; h[2] = (_Float16)a.z; h[3] = (_Float16)a.w;
    h[4] = (_Float16)b.x; h[5] = (_Float16)b.y; h[6] = (_Float16)b.z; h[7] = (_Float16)b.w;
    return h;
}

// ---- vectorized tile loaders (256 threads cooperatively) ----
// A tile 128x32 from f32 source (row-major, leading dim lda): 2 threads/row.
__device__ __forceinline__ void load_tileA_f32(const float* __restrict__ src, int lda,
                                               int m0, int k0, _Float16* As, int tid) {
    const int r = tid >> 1, c0 = (tid & 1) * 16;
    const float4* p = (const float4*)(src + (size_t)(m0 + r) * lda + k0 + c0);
    float4 f0 = p[0], f1 = p[1], f2 = p[2], f3 = p[3];
    *(v8h*)(As + r * TSTR + c0)     = cvt8(f0, f1);
    *(v8h*)(As + r * TSTR + c0 + 8) = cvt8(f2, f3);
}

// A tile 128x32 from f16 source.
__device__ __forceinline__ void load_tileA_f16(const _Float16* __restrict__ src, int lda,
                                               int m0, int k0, _Float16* As, int tid) {
    const int r = tid >> 1, c0 = (tid & 1) * 16;
    const v8h* p = (const v8h*)(src + (size_t)(m0 + r) * lda + k0 + c0);
    v8h a = p[0], b = p[1];
    *(v8h*)(As + r * TSTR + c0)     = a;
    *(v8h*)(As + r * TSTR + c0 + 8) = b;
}

// B tile 64x32 from f16 source whose rows ARE Bt rows (K stored [t][dk]).
__device__ __forceinline__ void load_tileB_f16_direct(const _Float16* __restrict__ src, int ldb,
                                                      int n0, int k0, _Float16* Bs, int tid) {
    const int r = tid >> 2, c0 = (tid & 3) * 8;
    *(v8h*)(Bs + r * TSTR + c0) = *(const v8h*)(src + (size_t)(n0 + r) * ldb + k0 + c0);
}

// Bt[n][k] = src[k][n], f32 source. Coalesced 128-bit global reads along n,
// scalar LDS scatter (transpose).
__device__ __forceinline__ void load_tileBt_f32(const float* __restrict__ src, int ldb,
                                                int n0, int k0, _Float16* Bs, int tid) {
    const int k = tid >> 3, nb = (tid & 7) * 8;
    const float4* p = (const float4*)(src + (size_t)(k0 + k) * ldb + n0 + nb);
    float4 f0 = p[0], f1 = p[1];
    v8h h = cvt8(f0, f1);
#pragma unroll
    for (int j = 0; j < 8; ++j) Bs[(nb + j) * TSTR + k] = h[j];
}

// Bt[n][k] = src[k][n], f16 source. One 128-bit global read per thread.
__device__ __forceinline__ void load_tileBt_f16(const _Float16* __restrict__ src, int ldb,
                                                int n0, int k0, _Float16* Bs, int tid) {
    const int k = tid >> 3, nb = (tid & 7) * 8;
    v8h v = *(const v8h*)(src + (size_t)(k0 + k) * ldb + n0 + nb);
#pragma unroll
    for (int j = 0; j < 8; ++j) Bs[(nb + j) * TSTR + k] = v[j];
}

// ------------------------------------------------------------------
// 1) Per-head projection: Out[h][row][n] = X[row][:] . W[h][:][n] + bias[h][n]
//    X: [8192,1024] f32, W: [H,1024,256] f32, Out: [H,8192,256] f16
//    grid: (8192/128, 256/64, H), block 256
// ------------------------------------------------------------------
__global__ __launch_bounds__(256)
void mha_proj_kernel(const float* __restrict__ X, const float* __restrict__ W,
                     const float* __restrict__ bias, _Float16* __restrict__ Out) {
    const int h  = blockIdx.z;
    const int m0 = blockIdx.x * 128;
    const int n0 = blockIdx.y * 64;
    __shared__ __align__(16) _Float16 As[128 * TSTR];
    __shared__ __align__(16) _Float16 Bs[64 * TSTR];
    const int tid  = threadIdx.x;
    const int lane = tid & 31;
    const int wid  = tid >> 5;
    const int wm   = (wid & 3) << 5;   // 0,32,64,96
    const int wn   = (wid >> 2) << 5;  // 0,32
    v8f acc[2][2] = {};
    const float* Wh = W + (size_t)h * DM_ * DK_;

    for (int k0 = 0; k0 < DM_; k0 += 32) {
        __syncthreads();
        load_tileA_f32(X, DM_, m0, k0, As, tid);
        load_tileBt_f32(Wh, DK_, n0, k0, Bs, tid);
        if (k0 + 32 < DM_) {   // prefetch next k-tile (global_prefetch_b8)
            __builtin_prefetch(X + (size_t)(m0 + (tid >> 1)) * DM_ + k0 + 32, 0, 3);
            __builtin_prefetch(Wh + (size_t)(k0 + 32 + (tid >> 3)) * DK_ + n0, 0, 3);
        }
        __syncthreads();
        v16h a0 = load_frag_a(As, wm, lane);
        v16h a1 = load_frag_a(As, wm + 16, lane);
        v16h b0 = load_frag_b(Bs, wn, lane);
        v16h b1 = load_frag_b(Bs, wn + 16, lane);
        acc[0][0] = WMMA_F16(a0, b0, acc[0][0]);
        acc[0][1] = WMMA_F16(a0, b1, acc[0][1]);
        acc[1][0] = WMMA_F16(a1, b0, acc[1][0]);
        acc[1][1] = WMMA_F16(a1, b1, acc[1][1]);
    }
    const float* bh = bias + h * DK_;
    const int n  = lane & 15;
    const int mo = (lane & 16) ? 8 : 0;
#pragma unroll
    for (int i = 0; i < 2; ++i)
#pragma unroll
        for (int j = 0; j < 2; ++j) {
            int mbase = m0 + wm + i * 16, nbase = n0 + wn + j * 16;
            float bv = bh[nbase + n];
#pragma unroll
            for (int r = 0; r < 8; ++r)
                Out[((size_t)h * ROWS_ + (mbase + mo + r)) * DK_ + nbase + n] =
                    (_Float16)(acc[i][j][r] + bv);
        }
}

// ------------------------------------------------------------------
// 2) Scores: attn[hb][q][t] = (Qh[q][:] . Kh[t][:]) * 1/16
//    K stored [t][dk] == Bt layout directly (no transpose needed).
//    grid: (4096/128, 4096/64, H*B)
// ------------------------------------------------------------------
__global__ __launch_bounds__(256)
void mha_scores_kernel(const _Float16* __restrict__ Q, const _Float16* __restrict__ K,
                       float* __restrict__ attn) {
    const int hb = blockIdx.z;           // h*B + b
    const int h = hb >> 1, b = hb & 1;
    const _Float16* Qh = Q + ((size_t)h * ROWS_ + (size_t)b * S_) * DK_;
    const _Float16* Kh = K + ((size_t)h * ROWS_ + (size_t)b * S_) * DK_;
    float* Ch = attn + (size_t)hb * S_ * S_;
    const int m0 = blockIdx.x * 128;
    const int n0 = blockIdx.y * 64;
    __shared__ __align__(16) _Float16 As[128 * TSTR];
    __shared__ __align__(16) _Float16 Bs[64 * TSTR];
    const int tid = threadIdx.x, lane = tid & 31, wid = tid >> 5;
    const int wm = (wid & 3) << 5, wn = (wid >> 2) << 5;
    v8f acc[2][2] = {};

    for (int k0 = 0; k0 < DK_; k0 += 32) {
        __syncthreads();
        load_tileA_f16(Qh, DK_, m0, k0, As, tid);
        load_tileB_f16_direct(Kh, DK_, n0, k0, Bs, tid);
        __syncthreads();
        v16h a0 = load_frag_a(As, wm, lane);
        v16h a1 = load_frag_a(As, wm + 16, lane);
        v16h b0 = load_frag_b(Bs, wn, lane);
        v16h b1 = load_frag_b(Bs, wn + 16, lane);
        acc[0][0] = WMMA_F16(a0, b0, acc[0][0]);
        acc[0][1] = WMMA_F16(a0, b1, acc[0][1]);
        acc[1][0] = WMMA_F16(a1, b0, acc[1][0]);
        acc[1][1] = WMMA_F16(a1, b1, acc[1][1]);
    }
    const float scale = 0.0625f;   // 1/sqrt(256)
    const int n = lane & 15;
    const int mo = (lane & 16) ? 8 : 0;
#pragma unroll
    for (int i = 0; i < 2; ++i)
#pragma unroll
        for (int j = 0; j < 2; ++j) {
            int mbase = m0 + wm + i * 16, nbase = n0 + wn + j * 16;
#pragma unroll
            for (int r = 0; r < 8; ++r)
                Ch[(size_t)(mbase + mo + r) * S_ + nbase + n] = acc[i][j][r] * scale;
        }
}

// ------------------------------------------------------------------
// 3) In-place row softmax over 4096 columns. One block per row.
//    float4 I/O: 4 x b128 loads / stores per thread.
// ------------------------------------------------------------------
__global__ __launch_bounds__(256)
void mha_softmax_kernel(float* __restrict__ attn) {
    float4* p = (float4*)(attn + (size_t)blockIdx.x * S_);
    const int tid = threadIdx.x;
    __shared__ float red[256];
    float4 v[4];
    float m = -3.0e38f;
#pragma unroll
    for (int i = 0; i < 4; ++i) {
        v[i] = p[tid + i * 256];
        m = fmaxf(m, fmaxf(fmaxf(v[i].x, v[i].y), fmaxf(v[i].z, v[i].w)));
    }
    red[tid] = m; __syncthreads();
    for (int s = 128; s > 0; s >>= 1) {
        if (tid < s) red[tid] = fmaxf(red[tid], red[tid + s]);
        __syncthreads();
    }
    m = red[0]; __syncthreads();
    float sum = 0.f;
#pragma unroll
    for (int i = 0; i < 4; ++i) {
        v[i].x = __expf(v[i].x - m); v[i].y = __expf(v[i].y - m);
        v[i].z = __expf(v[i].z - m); v[i].w = __expf(v[i].w - m);
        sum += v[i].x + v[i].y + v[i].z + v[i].w;
    }
    red[tid] = sum; __syncthreads();
    for (int s = 128; s > 0; s >>= 1) {
        if (tid < s) red[tid] += red[tid + s];
        __syncthreads();
    }
    const float inv = 1.0f / red[0];
#pragma unroll
    for (int i = 0; i < 4; ++i) {
        v[i].x *= inv; v[i].y *= inv; v[i].z *= inv; v[i].w *= inv;
        p[tid + i * 256] = v[i];
    }
}

// ------------------------------------------------------------------
// 4) head_out = attn @ V, written as f16 "concat" [8192][1024] with head
//    interleave: concat[(b*4096+q)*1024 + h*256 + dk]
//    grid: (4096/128, 256/64, H*B)
// ------------------------------------------------------------------
__global__ __launch_bounds__(256)
void mha_av_kernel(const float* __restrict__ attn, const _Float16* __restrict__ V,
                   _Float16* __restrict__ concat) {
    const int hb = blockIdx.z;
    const int h = hb >> 1, b = hb & 1;
    const float* Ah = attn + (size_t)hb * S_ * S_;                       // [q][s]
    const _Float16* Vh = V + ((size_t)h * ROWS_ + (size_t)b * S_) * DK_; // [s][dk]
    const int m0 = blockIdx.x * 128;   // q
    const int n0 = blockIdx.y * 64;    // dk
    __shared__ __align__(16) _Float16 As[128 * TSTR];
    __shared__ __align__(16) _Float16 Bs[64 * TSTR];
    const int tid = threadIdx.x, lane = tid & 31, wid = tid >> 5;
    const int wm = (wid & 3) << 5, wn = (wid >> 2) << 5;
    v8f acc[2][2] = {};

    for (int k0 = 0; k0 < S_; k0 += 32) {
        __syncthreads();
        load_tileA_f32(Ah, S_, m0, k0, As, tid);      // probs f32 -> f16
        load_tileBt_f16(Vh, DK_, n0, k0, Bs, tid);    // Bt[dk][s] = V[s][dk]
        if (k0 + 32 < S_) {
            __builtin_prefetch(Ah + (size_t)(m0 + (tid >> 1)) * S_ + k0 + 32, 0, 3);
            __builtin_prefetch(Vh + (size_t)(k0 + 32 + (tid >> 3)) * DK_ + n0, 0, 3);
        }
        __syncthreads();
        v16h a0 = load_frag_a(As, wm, lane);
        v16h a1 = load_frag_a(As, wm + 16, lane);
        v16h b0 = load_frag_b(Bs, wn, lane);
        v16h b1 = load_frag_b(Bs, wn + 16, lane);
        acc[0][0] = WMMA_F16(a0, b0, acc[0][0]);
        acc[0][1] = WMMA_F16(a0, b1, acc[0][1]);
        acc[1][0] = WMMA_F16(a1, b0, acc[1][0]);
        acc[1][1] = WMMA_F16(a1, b1, acc[1][1]);
    }
    const int n = lane & 15;
    const int mo = (lane & 16) ? 8 : 0;
#pragma unroll
    for (int i = 0; i < 2; ++i)
#pragma unroll
        for (int j = 0; j < 2; ++j) {
            int mbase = m0 + wm + i * 16, nbase = n0 + wn + j * 16;
#pragma unroll
            for (int r = 0; r < 8; ++r) {
                int q = mbase + mo + r;
                concat[((size_t)b * S_ + q) * DM_ + h * DK_ + nbase + n] =
                    (_Float16)acc[i][j][r];
            }
        }
}

// ------------------------------------------------------------------
// 5) out = concat @ Wo + bo   ([8192,1024] x [1024,1024] -> f32)
//    grid: (8192/128, 1024/64)
// ------------------------------------------------------------------
__global__ __launch_bounds__(256)
void mha_outproj_kernel(const _Float16* __restrict__ concat, const float* __restrict__ Wo,
                        const float* __restrict__ bo, float* __restrict__ out) {
    const int m0 = blockIdx.x * 128;
    const int n0 = blockIdx.y * 64;
    __shared__ __align__(16) _Float16 As[128 * TSTR];
    __shared__ __align__(16) _Float16 Bs[64 * TSTR];
    const int tid = threadIdx.x, lane = tid & 31, wid = tid >> 5;
    const int wm = (wid & 3) << 5, wn = (wid >> 2) << 5;
    v8f acc[2][2] = {};

    for (int k0 = 0; k0 < DM_; k0 += 32) {
        __syncthreads();
        load_tileA_f16(concat, DM_, m0, k0, As, tid);
        load_tileBt_f32(Wo, DM_, n0, k0, Bs, tid);    // Bt[n][k] = Wo[k][n]
        if (k0 + 32 < DM_) {
            __builtin_prefetch(concat + (size_t)(m0 + (tid >> 1)) * DM_ + k0 + 32, 0, 3);
            __builtin_prefetch(Wo + (size_t)(k0 + 32 + (tid >> 3)) * DM_ + n0, 0, 3);
        }
        __syncthreads();
        v16h a0 = load_frag_a(As, wm, lane);
        v16h a1 = load_frag_a(As, wm + 16, lane);
        v16h b0 = load_frag_b(Bs, wn, lane);
        v16h b1 = load_frag_b(Bs, wn + 16, lane);
        acc[0][0] = WMMA_F16(a0, b0, acc[0][0]);
        acc[0][1] = WMMA_F16(a0, b1, acc[0][1]);
        acc[1][0] = WMMA_F16(a1, b0, acc[1][0]);
        acc[1][1] = WMMA_F16(a1, b1, acc[1][1]);
    }
    const int n = lane & 15;
    const int mo = (lane & 16) ? 8 : 0;
#pragma unroll
    for (int i = 0; i < 2; ++i)
#pragma unroll
        for (int j = 0; j < 2; ++j) {
            int mbase = m0 + wm + i * 16, nbase = n0 + wn + j * 16;
            float bv = bo[nbase + n];
#pragma unroll
            for (int r = 0; r < 8; ++r)
                out[(size_t)(mbase + mo + r) * DM_ + nbase + n] = acc[i][j][r] + bv;
        }
}

// ------------------------------------------------------------------
extern "C" void kernel_launch(void* const* d_in, const int* in_sizes, int n_in,
                              void* d_out, int out_size, void* d_ws, size_t ws_size,
                              hipStream_t stream) {
    (void)in_sizes; (void)n_in; (void)out_size; (void)ws_size;
    const float* query = (const float*)d_in[0];
    const float* key   = (const float*)d_in[1];
    const float* value = (const float*)d_in[2];
    const float* Wq    = (const float*)d_in[3];
    const float* bq    = (const float*)d_in[4];
    const float* Wk    = (const float*)d_in[5];
    const float* bk    = (const float*)d_in[6];
    const float* Wv    = (const float*)d_in[7];
    const float* bv    = (const float*)d_in[8];
    const float* Wo    = (const float*)d_in[9];
    const float* bo    = (const float*)d_in[10];

    float* out  = (float*)d_out;                       // [2,4096,1024]
    float* attn = (float*)d_out + (size_t)ROWS_ * DM_; // [H,B,4096,4096]

    // workspace: f16 Qh, Kh, Vh ([H,8192,256]) and concat ([8192,1024])
    const size_t seg = (size_t)H_ * ROWS_ * DK_;       // 8,388,608 halves each
    _Float16* Qh     = (_Float16*)d_ws;
    _Float16* Kh     = Qh + seg;
    _Float16* Vh     = Kh + seg;
    _Float16* concat = Vh + seg;

    dim3 blk(256);
    dim3 gproj(ROWS_ / 128, DK_ / 64, H_);
    mha_proj_kernel<<<gproj, blk, 0, stream>>>(query, Wq, bq, Qh);
    mha_proj_kernel<<<gproj, blk, 0, stream>>>(key,   Wk, bk, Kh);
    mha_proj_kernel<<<gproj, blk, 0, stream>>>(value, Wv, bv, Vh);

    mha_scores_kernel<<<dim3(S_ / 128, S_ / 64, H_ * B_), blk, 0, stream>>>(Qh, Kh, attn);

    mha_softmax_kernel<<<dim3(H_ * B_ * S_), blk, 0, stream>>>(attn);

    mha_av_kernel<<<dim3(S_ / 128, DK_ / 64, H_ * B_), blk, 0, stream>>>(attn, Vh, concat);

    mha_outproj_kernel<<<dim3(ROWS_ / 128, DM_ / 64), blk, 0, stream>>>(concat, Wo, bo, out);
}
